// SelfAttnMatch_2327872274936
// MI455X (gfx1250) — matrix operand order repaired and checked by reading, not compiled, not run on hardware
//
#include <hip/hip_runtime.h>

typedef __bf16 bf16_t;
typedef __attribute__((ext_vector_type(16))) __bf16 v16bf;
typedef __attribute__((ext_vector_type(8)))  __bf16 v8bf;
typedef __attribute__((ext_vector_type(4)))  __bf16 v4bf;
typedef __attribute__((ext_vector_type(8)))  float  v8f;

#define NEG_INF (-__builtin_inff())

// ---------------- WMMA fragment builders (wave32, 16x16x32 bf16) -------------
// A (16x32 MxK): lanes 0-15 -> M=lane, K=0..15; lanes 16-31 -> M=lane-16, K=16..31
// Pointers passed in are already lane-adjusted (row + k0 applied).
__device__ __forceinline__ v16bf frag_a_from_bf16(const bf16_t* p) {
    v8bf lo = *(const v8bf*)(p);
    v8bf hi = *(const v8bf*)(p + 8);
    v16bf a;
#pragma unroll
    for (int i = 0; i < 8; ++i) { a[i] = lo[i]; a[8 + i] = hi[i]; }
    return a;
}

__device__ __forceinline__ v16bf frag_a_from_f32(const float* p) {
    v16bf a;
#pragma unroll
    for (int i = 0; i < 8; ++i) { a[i] = (bf16_t)p[i]; a[8 + i] = (bf16_t)p[8 + i]; }
    return a;
}

// B (32x16 KxN): lane n (0-15) holds B[0..15][n]; lane n+16 holds B[16..31][n].
__device__ __forceinline__ v16bf frag_b(const bf16_t* p) {
    return *(const v16bf*)(p);
}

#define WMMA_BF16(A, Bm, C) \
    __builtin_amdgcn_wmma_f32_16x16x32_bf16(false, (A), false, (Bm), (short)0, (C), false, false)

// ------------------------- kernel 0: f32 -> bf16 elementwise -----------------
__global__ void k_cvt(const float* __restrict__ in, bf16_t* __restrict__ outp, int n4) {
    const int i = (blockIdx.x * blockDim.x + threadIdx.x);
    if (i < n4) {
        const float4 v = *(const float4*)(in + (size_t)i * 4);
        v4bf o;
        o[0] = (bf16_t)v.x; o[1] = (bf16_t)v.y; o[2] = (bf16_t)v.z; o[3] = (bf16_t)v.w;
        *(v4bf*)(outp + (size_t)i * 4) = o;
    }
}

// ------------------------- kernel 1: x^T -> bf16 [B,D,L] ---------------------
__global__ void k_transpose(const float* __restrict__ x, bf16_t* __restrict__ xt,
                            int B, int L, int D) {
    __shared__ bf16_t tile[32][33];
    const int b  = blockIdx.z;
    const int l0 = blockIdx.y * 32;
    const int d0 = blockIdx.x * 32;
    const int tx = threadIdx.x;      // 0..31
    const int ty = threadIdx.y;      // 0..7
    const float* src = x + (size_t)b * L * D;
#pragma unroll
    for (int i = ty; i < 32; i += 8)
        tile[i][tx] = (bf16_t)src[(size_t)(l0 + i) * D + (d0 + tx)];
    __syncthreads();
    bf16_t* dst = xt + (size_t)b * D * L;
#pragma unroll
    for (int i = ty; i < 32; i += 8)
        dst[(size_t)(d0 + i) * L + (l0 + tx)] = tile[tx][i];
}

// ---------------- kernel 2: proj = relu(x @ W^T + b) -> bf16 -----------------
// One wave computes a 16x64 output strip (4 N-tiles, A reused 4x).
__global__ void __launch_bounds__(256) k_proj(const bf16_t* __restrict__ xb,
                                              const bf16_t* __restrict__ Wb,
                                              const float* __restrict__ bias,
                                              bf16_t* __restrict__ proj,
                                              int BL, int D) {
    const int lane = threadIdx.x & 31;
    const int wave = threadIdx.x >> 5;
    const int g    = blockIdx.x * 8 + wave;
    const int cgroups = D >> 6;              // groups of 4 N-tiles
    const int rt = g / cgroups;
    const int cg = g % cgroups;
    if (rt * 16 >= BL) return;

    const int n   = lane & 15;
    const int k0a = (lane & 16) ? 16 : 0;
    const int k0b = (lane & 16) ? 16 : 0;

    const bf16_t* ap = xb + ((size_t)rt * 16 + n) * D + k0a;
    const bf16_t* bp = Wb + ((size_t)cg * 64 + n) * D + k0b;

    v8f acc[4] = {};
    for (int dk = 0; dk < D; dk += 32) {
        const v16bf a = frag_a_from_bf16(ap); ap += 32;
        __builtin_prefetch(bp + 256, 0, 3);
#pragma unroll
        for (int c = 0; c < 4; ++c) {
            const v16bf bm = frag_b(bp + (size_t)c * 16 * D);
            acc[c] = WMMA_BF16(a, bm, acc[c]);
        }
        bp += 32;
    }

    // C layout: VGPR r, lanes 0-15 -> (M=r, N=lane); lanes 16-31 -> (M=8+r, N=lane-16)
    const int mo = (lane & 16) ? 8 : 0;
#pragma unroll
    for (int c = 0; c < 4; ++c) {
        const int e  = cg * 64 + c * 16 + n;
        const float bv = bias[e];
        bf16_t* orow = proj + (size_t)rt * 16 * D + e;
#pragma unroll
        for (int r = 0; r < 8; ++r) {
            float v = acc[c][r] + bv;
            v = v > 0.0f ? v : 0.0f;
            orow[(size_t)(r + mo) * D] = (bf16_t)v;
        }
    }
}

// -------- kernel 3: fused scores -> mask -> softmax -> P@x per 32-query strip --
// Block = 256 threads (8 waves). LDS holds the 32 x L fp32 score strip (256KB,
// legal on CDNA5's 320KB/WG). Two query tiles per block double B-operand reuse.
__global__ void __launch_bounds__(256) k_attn(const bf16_t* __restrict__ proj,
                                              const bf16_t* __restrict__ xt,
                                              const int* __restrict__ x_mask,
                                              float* __restrict__ out,
                                              int B, int L, int D) {
    extern __shared__ char smem_raw[];
    float* s_scores = (float*)smem_raw;                               // 32*L floats
    float* s_red    = (float*)(smem_raw + (size_t)32 * L * 4);        // 32*8 floats
    float* s_inv    = (float*)(smem_raw + (size_t)32 * L * 4 + 1024); // 32 floats

    const int tid  = threadIdx.x;
    const int lane = tid & 31;
    const int wave = tid >> 5;                 // 0..7
    const int qstrips = L >> 5;                // 64
    const int b     = blockIdx.x / qstrips;
    const int qbase = (blockIdx.x % qstrips) << 5;   // 32 queries per block

    const int n   = lane & 15;
    const int k0a = (lane & 16) ? 16 : 0;
    const int k0b = (lane & 16) ? 16 : 0;
    const int mo  = (lane & 16) ? 8 : 0;

    // ---- Phase A: score strip S[32 x L]; wave owns 256 keys (2 halves x 8 tiles).
    // Per K-step: 2 A-frags (2 query tiles) x 8 B-frags -> 16 WMMAs; B reused 2x, A 8x.
    const int j0w = wave * (L >> 3);
    const bf16_t* apq = proj + ((size_t)b * L + qbase + n) * D + k0a;
#pragma unroll 1
    for (int half = 0; half < 2; ++half) {
        const int j0h = j0w + half * 128;
        const bf16_t* ap0 = apq;
        const bf16_t* ap1 = apq + (size_t)16 * D;
        const bf16_t* bp  = proj + ((size_t)b * L + j0h + n) * D + k0b;
        v8f acc[2][8] = {};
        for (int dk = 0; dk < D; dk += 32) {
            const v16bf a0 = frag_a_from_bf16(ap0); ap0 += 32;
            const v16bf a1 = frag_a_from_bf16(ap1); ap1 += 32;
            __builtin_prefetch(bp + 256, 0, 3);
#pragma unroll
            for (int t = 0; t < 8; ++t) {
                const v16bf bm = frag_b(bp + (size_t)t * 16 * D);
                acc[0][t] = WMMA_BF16(a0, bm, acc[0][t]);
                acc[1][t] = WMMA_BF16(a1, bm, acc[1][t]);
            }
            bp += 32;
        }
#pragma unroll
        for (int t = 0; t < 8; ++t) {
            const int j = j0h + t * 16 + n;
            const bool pad = x_mask[b * L + j] != 0;
#pragma unroll
            for (int q = 0; q < 2; ++q) {
#pragma unroll
                for (int r = 0; r < 8; ++r) {
                    const int m = q * 16 + r + mo;   // row in 32-query strip
                    float s = acc[q][t][r];
                    if (j == qbase + m) s = 0.0f;    // zero diagonal
                    if (pad) s = NEG_INF;            // key padding mask (wins over diag)
                    s_scores[m * L + j] = s;
                }
            }
        }
    }
    __syncthreads();

    // ---- Phase B: softmax over keys (32 rows, 8 threads per row) ----
    {
        const int row = tid >> 3;
        const int t   = tid & 7;
        float mx = NEG_INF;
        for (int c = t; c < L; c += 8) mx = fmaxf(mx, s_scores[row * L + c]);
        s_red[row * 8 + t] = mx;
        __syncthreads();
        float rowmax = NEG_INF;
#pragma unroll
        for (int k = 0; k < 8; ++k) rowmax = fmaxf(rowmax, s_red[row * 8 + k]);
        __syncthreads();
        float sum = 0.0f;
        for (int c = t; c < L; c += 8) {
            float e = __expf(s_scores[row * L + c] - rowmax);
            s_scores[row * L + c] = e;               // keep unnormalized probs in place
            sum += e;
        }
        s_red[row * 8 + t] = sum;
        __syncthreads();
        float tot = 0.0f;
#pragma unroll
        for (int k = 0; k < 8; ++k) tot += s_red[row * 8 + k];
        if (t == 0) s_inv[row] = 1.0f / tot;         // fold normalization into phase C
    }
    __syncthreads();

    // ---- Phase C: out[32 x D] = (P / rowsum) @ x; wave owns 8 N-tiles,
    //      2 prob fragments (2 query tiles) share every B fragment.
    {
        const float* ap0 = s_scores + (size_t)n * L + k0a;        // probs, query tile 0
        const float* ap1 = s_scores + (size_t)(16 + n) * L + k0a; // probs, query tile 1
        const bf16_t* bp = xt + ((size_t)b * D + wave * 128 + n) * L + k0b;
        v8f acc[2][8] = {};
        for (int j0 = 0; j0 < L; j0 += 32) {
            const v16bf a0 = frag_a_from_f32(ap0 + j0);  // cvt to bf16, reused 8x
            const v16bf a1 = frag_a_from_f32(ap1 + j0);
#pragma unroll
            for (int c = 0; c < 8; ++c) {
                const v16bf bm = frag_b(bp + (size_t)c * 16 * L + j0);
                acc[0][c] = WMMA_BF16(a0, bm, acc[0][c]);
                acc[1][c] = WMMA_BF16(a1, bm, acc[1][c]);
            }
        }
#pragma unroll
        for (int q = 0; q < 2; ++q) {
            float inv[8];
#pragma unroll
            for (int r = 0; r < 8; ++r) inv[r] = s_inv[q * 16 + r + mo];
#pragma unroll
            for (int c = 0; c < 8; ++c) {
                float* orow = out + ((size_t)b * L + qbase + q * 16) * D
                            + (wave * 128 + c * 16) + n;
#pragma unroll
                for (int r = 0; r < 8; ++r)
                    orow[(size_t)(r + mo) * D] = acc[q][c][r] * inv[r];
            }
        }
    }
}

// ------------------------------- launcher ------------------------------------
extern "C" void kernel_launch(void* const* d_in, const int* in_sizes, int n_in,
                              void* d_out, int out_size, void* d_ws, size_t ws_size,
                              hipStream_t stream) {
    const float* x      = (const float*)d_in[0];
    const int*   x_mask = (const int*)d_in[1];
    const float* W      = (const float*)d_in[2];
    const float* bias   = (const float*)d_in[3];
    float*       out    = (float*)d_out;

    const int B = 8, L = 2048, D = 1024;
    const size_t nx = (size_t)B * L * D;           // 16.7M
    const size_t nw = (size_t)D * D;               // 1M

    bf16_t* proj = (bf16_t*)d_ws;                  // 32 MB
    bf16_t* xt   = proj + nx;                      // 32 MB
    bf16_t* xb   = xt + nx;                        // 32 MB (row-major x in bf16)
    bf16_t* Wb   = xb + nx;                        // 2 MB  (W in bf16)

    // 0) f32 -> bf16 copies of x (row-major) and W
    k_cvt<<<(int)(nx / 4 + 255) / 256, 256, 0, stream>>>(x, xb, (int)(nx / 4));
    k_cvt<<<(int)(nw / 4 + 255) / 256, 256, 0, stream>>>(W, Wb, (int)(nw / 4));

    // 1) transpose x -> bf16 [B, D, L]
    dim3 tb(32, 8);
    dim3 tg(D / 32, L / 32, B);
    k_transpose<<<tg, tb, 0, stream>>>(x, xt, B, L, D);

    // 2) projection GEMM + bias + relu (wave = 16x64 strip)
    const int BL = B * L;
    const int waves  = (BL / 16) * (D / 64);       // 16384
    const int blocks = waves / 8;                  // 2048
    k_proj<<<blocks, 256, 0, stream>>>(xb, Wb, bias, proj, BL, D);

    // 3) fused scores + softmax + P@x (256KB LDS score strip per block)
    const size_t shmem = (size_t)32 * L * 4 + 1024 + 128; // 263,296 B (<320KB/WG on CDNA5)
    (void)hipFuncSetAttribute((const void*)k_attn,
                              hipFuncAttributeMaxDynamicSharedMemorySize,
                              (int)shmem);
    const int agrid = B * (L / 32);                // 512 blocks
    k_attn<<<agrid, 256, shmem, stream>>>(proj, xt, x_mask, out, B, L, D);
}
